// ResidualVQ_74706661147168
// MI455X (gfx1250) — matrix operand order repaired and checked by reading, not compiled, not run on hardware
//
#include <hip/hip_runtime.h>
#include <hip/hip_bf16.h>

// Problem constants (from reference): B=8, N=4096, D=512, C=1024, Q=8
#define Bv 8
#define Nv 4096
#define Dv 512
#define Cv 1024
#define Qv 8
#define Mv (Bv * Nv)          // 32768 rows
#define MT 128                // rows per block (8 waves x 16 rows)
#define APAD 8                // LDS row pad (halves): stride 1040 B -> no bank conflicts
#define AROW (Dv + APAD)      // 520 halves per padded LDS row
#define BUFH (16 * AROW)      // halves per B buffer

typedef __attribute__((ext_vector_type(16))) __bf16 bf16x16;
typedef __attribute__((ext_vector_type(8)))  float  f32x8;
typedef unsigned int u32x4 __attribute__((ext_vector_type(4)));
typedef int          i32x8 __attribute__((ext_vector_type(8)));
typedef int          i32x4 __attribute__((ext_vector_type(4)));

#if defined(__gfx1250__) && __has_builtin(__builtin_amdgcn_tensor_load_to_lds) && \
    __has_builtin(__builtin_amdgcn_s_wait_tensorcnt)
#define USE_TDM 1
#else
#define USE_TDM 0
#endif

__device__ __forceinline__ unsigned short f32_to_bf16_rne(float f) {
    unsigned u = __float_as_uint(f);
    u += 0x7FFFu + ((u >> 16) & 1u);       // round-to-nearest-even
    return (unsigned short)(u >> 16);
}

// ---------------------------------------------------------------------------
// init: residual = x (fp32 + bf16), qout = 0, losses = 0
// ---------------------------------------------------------------------------
__global__ void __launch_bounds__(256) init_kernel(
    const float* __restrict__ x,
    float* __restrict__ residual,
    unsigned short* __restrict__ rbf,
    float* __restrict__ qout,
    float* __restrict__ losses) {
    int i = blockIdx.x * 256 + threadIdx.x;      // < Mv*Dv
    float v = x[i];
    residual[i] = v;
    rbf[i] = f32_to_bf16_rne(v);
    qout[i] = 0.0f;
    if (i < Qv) losses[i] = 0.0f;
}

// ---------------------------------------------------------------------------
// prep: bf16 codebooks + per-code squared norms (one block per code row)
// ---------------------------------------------------------------------------
__global__ void __launch_bounds__(128) prep_codebooks(
    const float* __restrict__ cb,          // [Q*C, D] fp32
    unsigned short* __restrict__ cbbf,     // [Q*C, D] bf16
    float* __restrict__ cnorm) {           // [Q*C]
    int row = blockIdx.x;
    int t = threadIdx.x;                   // 128 threads = 4 waves
    const float* src = cb + (size_t)row * Dv;
    unsigned short* dst = cbbf + (size_t)row * Dv;
    float s = 0.0f;
    for (int d = t; d < Dv; d += 128) {
        float v = src[d];
        dst[d] = f32_to_bf16_rne(v);
        s += v * v;
    }
    #pragma unroll
    for (int off = 16; off >= 1; off >>= 1) s += __shfl_down(s, off);
    __shared__ float part[4];
    int wave = t >> 5, lane = t & 31;
    if (lane == 0) part[wave] = s;
    __syncthreads();
    if (t == 0) cnorm[row] = part[0] + part[1] + part[2] + part[3];
}

// ---------------------------------------------------------------------------
// Stage one 16x512 bf16 codebook tile into (padded) LDS.
// TDM path: 2-D tensor descriptor, data_size=2B, LDS padding 16B per 1024B
// (pad_interval=256 DWORDs -> code 7, pad_amount=4 DWORDs -> code 3) which
// reproduces the AROW=520-half padded row layout. Issued by wave 0 only.
// ---------------------------------------------------------------------------
__device__ __forceinline__ void stage_btile(unsigned short* dstbuf,
                                            const unsigned short* src,
                                            int tid, int wave) {
#if USE_TDM
    if (wave == 0) {
        unsigned long long ga = (unsigned long long)src;
        unsigned lds_off = (unsigned)(unsigned long long)(const void*)dstbuf;
        u32x4 g0;
        g0.x = 1u;                                        // count=1, user desc
        g0.y = lds_off;                                   // D#.lds_addr
        g0.z = (unsigned)ga;                              // global_addr[31:0]
        g0.w = (unsigned)(ga >> 32) | (2u << 30);         // addr[56:32] | type=2
        i32x8 g1;
        g1[0] = (1 << 16) | (1 << 20) | (7 << 22) | (3 << 25);
        //       data=2B    pad_en      intv=256DW  amt=4DW ; wg_mask=0
        g1[1] = (int)(512u << 16);        // tensor_dim0[15:0]=512 (barrier addr 0)
        g1[2] = (int)(1024u << 16);       // tensor_dim0 hi=0; tensor_dim1 lo=1024
        g1[3] = (int)(512u << 16);        // tensor_dim1 hi=0; tile_dim0=512
        g1[4] = 16;                       // tile_dim1=16, tile_dim2=0
        g1[5] = 512;                      // tensor_dim0_stride[31:0]=512 elems
        g1[6] = 0;                        // stride hi / tensor_dim1_stride lo
        g1[7] = 0;
        i32x4 z4 = {};
#if __clang_major__ >= 23
        i32x8 z8 = {};
        __builtin_amdgcn_tensor_load_to_lds(g0, g1, z4, z4, z8, 0);
#else
        __builtin_amdgcn_tensor_load_to_lds(g0, g1, z4, z4, 0);
#endif
    }
#else
    const uint4* s4 = (const uint4*)src;
    for (int i = tid; i < 16 * (Dv / 8); i += 256) {      // 1024 uint4
        int r = i >> 6;
        int c = i & 63;
        ((uint4*)(dstbuf + r * AROW))[c] = s4[r * 64 + c];
    }
#endif
}

__device__ __forceinline__ void stage_wait(int wave) {
#if USE_TDM
    if (wave == 0) __builtin_amdgcn_s_wait_tensorcnt(0);
#endif
}

// ---------------------------------------------------------------------------
// Fused score GEMM (bf16 WMMA, fp32 acc) + argmin.
// Block = 256 thr (8 waves) owns MT=128 rows; wave w owns rows w*16..w*16+15
// with its full 16x512 bf16 A-tile resident in VGPRs (32 uint4/lane).
// B (codebook) streamed through double-buffered LDS, shared by all 8 waves:
// codebook L2 traffic = (M/128) * 1MB = 256MB/stage -> ~120 FLOP/byte.
// score(m,c) = ||e_c||^2 - 2*(r_m . e_c)  (||r||^2 constant over c, dropped)
// ---------------------------------------------------------------------------
__global__ void __launch_bounds__(256, 1) argmin_kernel(
    const unsigned short* __restrict__ rbf,     // [M, D] bf16 residual
    const unsigned short* __restrict__ cbbf_q,  // [C, D] bf16 codebook, stage q
    const float* __restrict__ cnorm_q,          // [C]
    int* __restrict__ gidx,                     // [M] out
    float* __restrict__ idx_out,                // d_out indices region [M, Q]
    int q) {
    __shared__ unsigned short btile[2 * BUFH];  // double-buffered 16x512 B tile

    const int tid = threadIdx.x;
    const int wave = tid >> 5;
    const int lane = tid & 31;
    const int group = lane >> 4;                // 0: lanes 0-15, 1: lanes 16-31
    const int lane16 = lane & 15;
    const int rowbase = blockIdx.x * MT;
    const int myrow = rowbase + wave * 16 + lane16;   // A-matrix M index

    // --- Load this wave's A tile into registers (layout per ISA 7.12.2):
    // 16-bit A 16x32: halves 0-7 -> K=kb*32+8*group.., halves 8-15 -> +16
    union AF { uint4 u[2]; bf16x16 v; };
    AF afr[16];                                 // 128 VGPRs/lane, resident
    {
        const uint4* arowg = (const uint4*)(rbf + (size_t)myrow * Dv);
        #pragma unroll
        for (int kb = 0; kb < 16; ++kb) {
            afr[kb].u[0] = arowg[kb * 4 + group];
            afr[kb].u[1] = arowg[kb * 4 + 2 + group];
        }
    }

    float best_s[8];
    int   best_i[8];
    #pragma unroll
    for (int i = 0; i < 8; ++i) { best_s[i] = 3.4e38f; best_i[i] = 0x7fffffff; }

    // Prologue: stage tile 0
    stage_btile(btile, cbbf_q, tid, wave);
    stage_wait(wave);
    __syncthreads();

    #pragma unroll 1
    for (int ct = 0; ct < Cv / 16; ++ct) {      // 64 column tiles
        const int cur = ct & 1;
        if (ct + 1 < Cv / 16)                   // async prefetch next tile
            stage_btile(btile + (1 - cur) * BUFH,
                        cbbf_q + (size_t)(ct + 1) * 16 * Dv, tid, wave);

        // B 32x16 layout: lane=N(=lane16); halves 0-7 -> K 8*group.., halves 8-15 -> +16
        const uint4* b4 = (const uint4*)(btile + cur * BUFH + lane16 * AROW);
        f32x8 acc = {};
        #pragma unroll
        for (int kb = 0; kb < 16; ++kb) {
            union BF { uint4 u[2]; bf16x16 v; } b;
            const int bi = kb * 4 + group;
            b.u[0] = b4[bi];
            b.u[1] = b4[bi + 2];
            acc = __builtin_amdgcn_wmma_f32_16x16x32_bf16(
                false, afr[kb].v, false, b.v, (short)0, acc, false, false);
        }

        const int n = ct * 16 + lane16;
        const float cn = cnorm_q[n];
        #pragma unroll
        for (int i = 0; i < 8; ++i) {           // C/D: VGPR i -> M=i+8*group
            float s = cn - 2.0f * acc[i];
            if (s < best_s[i] || (s == best_s[i] && n < best_i[i])) {
                best_s[i] = s; best_i[i] = n;
            }
        }

        stage_wait(wave);                       // prefetch complete before publish
        __syncthreads();
    }

    // Argmin across the 16 N-lanes within each half-wave
    #pragma unroll
    for (int off = 8; off >= 1; off >>= 1) {
        #pragma unroll
        for (int i = 0; i < 8; ++i) {
            float os = __shfl_xor(best_s[i], off);
            int   oi = __shfl_xor(best_i[i], off);
            if (os < best_s[i] || (os == best_s[i] && oi < best_i[i])) {
                best_s[i] = os; best_i[i] = oi;
            }
        }
    }
    if (lane16 == 0) {                          // lanes 0 and 16: rows i+8*group
        #pragma unroll
        for (int i = 0; i < 8; ++i) {
            int m = rowbase + wave * 16 + i + 8 * group;
            gidx[m] = best_i[i];
            idx_out[(size_t)m * Qv + q] = (float)best_i[i];
        }
    }
}

// ---------------------------------------------------------------------------
// update: gather fp32 embed, residual -= e (fp32 + bf16), qout += e,
// loss[q] += mean((e - r)^2) via block reduction + one atomicAdd per block
// ---------------------------------------------------------------------------
__global__ void __launch_bounds__(256) update_kernel(
    const float* __restrict__ cb_q,        // [C, D] fp32, stage q
    const int* __restrict__ gidx,          // [M]
    float* __restrict__ residual,          // [M, D]
    unsigned short* __restrict__ rbf,      // [M, D]
    float* __restrict__ qout,              // [M, D] (in d_out)
    float* __restrict__ loss_ptr) {        // &losses[q]
    int i = blockIdx.x * 256 + threadIdx.x;      // < Mv*Dv
    int m = i >> 9;                              // / Dv
    int d = i & (Dv - 1);
    int idx = gidx[m];
    float e = cb_q[(size_t)idx * Dv + d];
    float r = residual[i];
    float diff = e - r;
    float nr = r - e;
    residual[i] = nr;
    rbf[i] = f32_to_bf16_rne(nr);
    qout[i] += e;

    float v = diff * diff;
    #pragma unroll
    for (int off = 16; off >= 1; off >>= 1) v += __shfl_down(v, off);
    __shared__ float part[8];
    int lane = threadIdx.x & 31, wave = threadIdx.x >> 5;
    if (lane == 0) part[wave] = v;
    __syncthreads();
    if (threadIdx.x == 0) {
        float s = 0.0f;
        #pragma unroll
        for (int w = 0; w < 8; ++w) s += part[w];
        atomicAdd(loss_ptr, s * (1.0f / (float)(Mv * Dv)));
    }
}

// ---------------------------------------------------------------------------
extern "C" void kernel_launch(void* const* d_in, const int* in_sizes, int n_in,
                              void* d_out, int out_size, void* d_ws, size_t ws_size,
                              hipStream_t stream) {
    const float* x  = (const float*)d_in[0];   // [B, N, D]
    const float* cb = (const float*)d_in[1];   // [Q, C, D]

    float* out     = (float*)d_out;
    float* qout    = out;                                   // M*D
    float* idx_out = out + (size_t)Mv * Dv;                 // M*Q
    float* losses  = idx_out + (size_t)Mv * Qv;             // Q

    char* ws = (char*)d_ws;
    float*          residual = (float*)ws;          ws += (size_t)Mv * Dv * 4;      // 64 MB
    unsigned short* rbf      = (unsigned short*)ws; ws += (size_t)Mv * Dv * 2;      // 32 MB
    unsigned short* cbbf     = (unsigned short*)ws; ws += (size_t)Qv * Cv * Dv * 2; // 8 MB
    float*          cnorm    = (float*)ws;          ws += (size_t)Qv * Cv * 4;      // 32 KB
    int*            gidx     = (int*)ws;            ws += (size_t)Mv * 4;           // 128 KB

    init_kernel<<<(Mv * Dv) / 256, 256, 0, stream>>>(x, residual, rbf, qout, losses);
    prep_codebooks<<<Qv * Cv, 128, 0, stream>>>(cb, cbbf, cnorm);

    for (int q = 0; q < Qv; ++q) {
        argmin_kernel<<<Mv / MT, 256, 0, stream>>>(
            rbf, cbbf + (size_t)q * Cv * Dv, cnorm + (size_t)q * Cv, gidx, idx_out, q);
        update_kernel<<<(Mv * Dv) / 256, 256, 0, stream>>>(
            cb + (size_t)q * Cv * Dv, gidx, residual, rbf, qout, losses + q);
    }
}
